// MHABlock_88888643158595
// MI455X (gfx1250) — matrix operand order repaired and checked by reading, not compiled, not run on hardware
//
#include <hip/hip_runtime.h>

// ---------------- types ----------------
typedef __attribute__((ext_vector_type(16))) __bf16 v16bf;
typedef __attribute__((ext_vector_type(8)))  float  v8f;
typedef unsigned int u32x4 __attribute__((ext_vector_type(4)));

union Frag16 { v16bf v; u32x4 q[2]; };

__device__ __forceinline__ unsigned short f2bf(float x) {
  unsigned int u = __float_as_uint(x);
  u = (u + 0x7FFFu + ((u >> 16) & 1u)) >> 16;
  return (unsigned short)u;
}

__device__ __forceinline__ v8f wmma_bf16(v16bf a, v16bf b, v8f c) {
  return __builtin_amdgcn_wmma_f32_16x16x32_bf16(false, a, false, b, (short)0, c,
                                                 false, false);
}

// A fragment (16x32, M x K), bf16, from row-major LDS [rows][ld]:
// lane l: m = l&15, g = l>>4; elems 0..15 -> K = k0+g*16 .. +15
__device__ __forceinline__ v16bf load_a_frag(const unsigned short* lds, int ld,
                                             int row0, int k0, int lane) {
  int g = lane >> 4, m = lane & 15;
  const unsigned short* p = lds + (size_t)(row0 + m) * ld + k0 + g * 16;
  Frag16 f;
  f.q[0] = *(const u32x4*)(p);
  f.q[1] = *(const u32x4*)(p + 8);
  return f.v;
}

// B fragment (32x16, K x N), bf16, from K-contiguous "Bt" LDS [N][ld over K]:
// lane l: n = l&15, g = l>>4; elems 0..15 -> K = k0+g*16 .. +15
__device__ __forceinline__ v16bf load_b_frag(const unsigned short* ldsT, int ld,
                                             int col0, int k0, int lane) {
  int g = lane >> 4, n = lane & 15;
  const unsigned short* p = ldsT + (size_t)(col0 + n) * ld + k0 + g * 16;
  Frag16 f;
  f.q[0] = *(const u32x4*)(p);
  f.q[1] = *(const u32x4*)(p + 8);
  return f.v;
}

// ---------------- kernel 1: Q/K/V GEMM (bf16 WMMA, f32 accum) ----------------
// Y[n,o] = sum_i X[n,i] * W[o,i];  X: [25600,640] f32, W: [640,640] f32.
// Block tile 128(M) x 64(N), K chunk 64. 256 threads = 8 waves, wave w owns
// rows [w*16, w*16+16). Output stored bf16, natural [25600,640] layout.
__global__ __launch_bounds__(256) void qkv_gemm_kernel(
    const float* __restrict__ X, const float* __restrict__ Wq,
    const float* __restrict__ Wk, const float* __restrict__ Wv,
    unsigned short* __restrict__ Qb, unsigned short* __restrict__ Kb,
    unsigned short* __restrict__ Vb) {
  __shared__ unsigned short sX[128 * 72];
  __shared__ unsigned short sW[64 * 72];

  const int t = threadIdx.x, lane = t & 31, wv = t >> 5;
  const int row0 = blockIdx.x * 128;
  const int n0 = blockIdx.y * 64;
  const int z = blockIdx.z;
  const float* W = (z == 0) ? Wq : ((z == 1) ? Wk : Wv);
  unsigned short* Out = (z == 0) ? Qb : ((z == 1) ? Kb : Vb);

  v8f acc[4];
#pragma unroll
  for (int nt = 0; nt < 4; ++nt)
#pragma unroll
    for (int e = 0; e < 8; ++e) acc[nt][e] = 0.f;

  for (int k0 = 0; k0 < 640; k0 += 64) {
    // stage X chunk [128][64] and W chunk (Bt layout) [64][64]
#pragma unroll 4
    for (int i = 0; i < 32; ++i) {
      int idx = t + i * 256;
      int r = idx >> 6, c = idx & 63;
      sX[r * 72 + c] = f2bf(X[(size_t)(row0 + r) * 640 + k0 + c]);
    }
#pragma unroll 4
    for (int i = 0; i < 16; ++i) {
      int idx = t + i * 256;
      int r = idx >> 6, c = idx & 63;
      sW[r * 72 + c] = f2bf(W[(size_t)(n0 + r) * 640 + k0 + c]);
    }
    __syncthreads();

#pragma unroll
    for (int kf = 0; kf < 2; ++kf) {
      v16bf a = load_a_frag(sX, 72, wv * 16, kf * 32, lane);
#pragma unroll
      for (int nt = 0; nt < 4; ++nt) {
        v16bf b = load_b_frag(sW, 72, nt * 16, kf * 32, lane);
        acc[nt] = wmma_bf16(a, b, acc[nt]);
      }
    }
    __syncthreads();
  }

  // store bf16 (C layout: lane covers col n = lane&15, rows r + (lane>>4)*8)
  const int g = lane >> 4, cn = lane & 15;
#pragma unroll
  for (int nt = 0; nt < 4; ++nt)
#pragma unroll
    for (int r = 0; r < 8; ++r)
      Out[(size_t)(row0 + wv * 16 + g * 8 + r) * 640 + n0 + nt * 16 + cn] =
          f2bf(acc[nt][r]);
}

// ---------------- kernel 2: fused attention + residual + LayerNorm ----------
// One block per (b,h). SEQ=400, dk=64. Two-pass softmax, WMMA throughout.
#define LDQ 72
#define LDV 416
#define LDP 104
#define SCALE_QK 0.125f
#define L2E 1.44269504088896f

__global__ __launch_bounds__(256) void attn_ln_kernel(
    const unsigned short* __restrict__ Qb, const unsigned short* __restrict__ Kb,
    const unsigned short* __restrict__ Vb, const float* __restrict__ X,
    const float* __restrict__ gamma, const float* __restrict__ beta,
    float* __restrict__ out) {
  extern __shared__ char smem[];
  unsigned short* sQ = (unsigned short*)smem;          // [400][72]
  unsigned short* sKt = sQ + 400 * LDQ;                // [400][72]  Kt[key][d]
  unsigned short* sVt = sKt + 400 * LDQ;               // [64][416]  Vt[d][key]
  unsigned short* sP = sVt + 64 * LDV;                 // [400][104] P[row][key]
  float* sM = (float*)(sP + 400 * LDP);                // [400] row max
  float* sL = sM + 400;                                // [400] 1/row sumexp
  float* obuf = (float*)smem;                          // overlay: [40][640] f32

  const int bh = blockIdx.x;
  const int b = bh / 10, h = bh % 10;
  const int t = threadIdx.x, lane = t & 31, wv = t >> 5;
  const int g = lane >> 4, cn = lane & 15;
  const size_t base = (size_t)(b * 400) * 640 + (size_t)h * 64;

  // ---- stage Q, Kt (raw-reshape transpose), Vt ----
#pragma unroll 4
  for (int i = 0; i < 100; ++i) {
    int f = t + i * 256;              // f = s*64 + d, 0..25599
    int s = f >> 6, d = f & 63;
    size_t gidx = base + (size_t)s * 640 + d;
    sQ[s * LDQ + d] = Qb[gidx];
    sKt[(f % 400) * LDQ + (f / 400)] = Kb[gidx];  // Kt[key][dd] = Kmem[dd*400+key]
    sVt[d * LDV + s] = Vb[gidx];
  }
  for (int i = t; i < 400 * LDP; i += 256) sP[i] = 0;     // incl. K-pad cols
  for (int i = t; i < 64 * 16; i += 256)                  // zero Vt pad keys 400..415
    sVt[(i >> 4) * LDV + 400 + (i & 15)] = 0;
  __syncthreads();

  // ---- pass 1: row max & sum-exp (online, per owned 16-row tile) ----
  for (int tt = 0; tt < 4; ++tt) {
    int mt = wv + tt * 8;
    if (mt >= 25) break;
    v16bf a0 = load_a_frag(sQ, LDQ, mt * 16, 0, lane);
    v16bf a1 = load_a_frag(sQ, LDQ, mt * 16, 32, lane);
    float mr[8], lr[8];
#pragma unroll
    for (int r = 0; r < 8; ++r) { mr[r] = -1e30f; lr[r] = 0.f; }
    for (int kt = 0; kt < 25; ++kt) {
      v16bf b0 = load_b_frag(sKt, LDQ, kt * 16, 0, lane);
      v16bf b1 = load_b_frag(sKt, LDQ, kt * 16, 32, lane);
      v8f c;
#pragma unroll
      for (int e = 0; e < 8; ++e) c[e] = 0.f;
      c = wmma_bf16(a0, b0, c);
      c = wmma_bf16(a1, b1, c);
#pragma unroll
      for (int r = 0; r < 8; ++r) {
        float v = c[r] * SCALE_QK;
        float mo = mr[r];
        float mn = fmaxf(mo, v);
        lr[r] = lr[r] * exp2f((mo - mn) * L2E) + exp2f((v - mn) * L2E);
        mr[r] = mn;
      }
    }
    // reduce across the 16 lanes of each half-wave
#pragma unroll
    for (int r = 0; r < 8; ++r) {
      for (int off = 1; off < 16; off <<= 1) {
        float mo = __shfl_xor(mr[r], off, 32);
        float lo = __shfl_xor(lr[r], off, 32);
        float mn = fmaxf(mr[r], mo);
        lr[r] = lr[r] * exp2f((mr[r] - mn) * L2E) + lo * exp2f((mo - mn) * L2E);
        mr[r] = mn;
      }
    }
    if (cn == 0) {
#pragma unroll
      for (int r = 0; r < 8; ++r) {
        int row = mt * 16 + g * 8 + r;
        sM[row] = mr[r];
        sL[row] = 1.f / lr[r];
      }
    }
  }
  __syncthreads();

  // ---- pass 2: recompute scores in 80-key chunks, P->LDS, O += P@V ----
  v8f acc[4][4];
#pragma unroll
  for (int a = 0; a < 4; ++a)
#pragma unroll
    for (int nt = 0; nt < 4; ++nt)
#pragma unroll
      for (int e = 0; e < 8; ++e) acc[a][nt][e] = 0.f;

  for (int ch = 0; ch < 5; ++ch) {
    for (int tt = 0; tt < 4; ++tt) {
      int mt = wv + tt * 8;
      if (mt >= 25) break;
      v16bf a0 = load_a_frag(sQ, LDQ, mt * 16, 0, lane);
      v16bf a1 = load_a_frag(sQ, LDQ, mt * 16, 32, lane);
      float mrow[8], rl[8];
#pragma unroll
      for (int r = 0; r < 8; ++r) {
        int row = mt * 16 + g * 8 + r;
        mrow[r] = sM[row];
        rl[r] = sL[row];
      }
#pragma unroll
      for (int ktl = 0; ktl < 5; ++ktl) {
        int kt = ch * 5 + ktl;
        v16bf b0 = load_b_frag(sKt, LDQ, kt * 16, 0, lane);
        v16bf b1 = load_b_frag(sKt, LDQ, kt * 16, 32, lane);
        v8f c;
#pragma unroll
        for (int e = 0; e < 8; ++e) c[e] = 0.f;
        c = wmma_bf16(a0, b0, c);
        c = wmma_bf16(a1, b1, c);
#pragma unroll
        for (int r = 0; r < 8; ++r) {
          float p = exp2f((c[r] * SCALE_QK - mrow[r]) * L2E) * rl[r];
          sP[(size_t)(mt * 16 + g * 8 + r) * LDP + ktl * 16 + cn] = f2bf(p);
        }
      }
    }
    __syncthreads();

    for (int tt = 0; tt < 4; ++tt) {
      int mt = wv + tt * 8;
      if (mt >= 25) break;
      v16bf p0 = load_a_frag(sP, LDP, mt * 16, 0, lane);
      v16bf p1 = load_a_frag(sP, LDP, mt * 16, 32, lane);
      v16bf p2 = load_a_frag(sP, LDP, mt * 16, 64, lane);  // keys 80..95 are zero pad
#pragma unroll
      for (int nt = 0; nt < 4; ++nt) {
        v16bf b0 = load_b_frag(sVt, LDV, nt * 16, ch * 80, lane);
        v16bf b1 = load_b_frag(sVt, LDV, nt * 16, ch * 80 + 32, lane);
        v16bf b2 = load_b_frag(sVt, LDV, nt * 16, ch * 80 + 64, lane);
        acc[tt][nt] = wmma_bf16(p0, b0, acc[tt][nt]);
        acc[tt][nt] = wmma_bf16(p1, b1, acc[tt][nt]);
        acc[tt][nt] = wmma_bf16(p2, b2, acc[tt][nt]);
      }
    }
    __syncthreads();
  }

  // ---- scatter O through the raw-reshape mapping into obuf [40][640] ----
  for (int tt = 0; tt < 4; ++tt) {
    int mt = wv + tt * 8;
    if (mt >= 25) break;
#pragma unroll
    for (int nt = 0; nt < 4; ++nt)
#pragma unroll
      for (int r = 0; r < 8; ++r) {
        int q = mt * 16 + g * 8 + r;
        int d = nt * 16 + cn;
        obuf[(q / 10) * 640 + (q % 10) * 64 + d] = acc[tt][nt][r];
      }
  }
  __syncthreads();

  // ---- residual + LayerNorm over the 40 output rows owned by this block ----
  const int nb0 = b * 400 + h * 40;
  for (int j = wv * 5; j < wv * 5 + 5; ++j) {
    int n = nb0 + j;
    float vbuf[20], sum = 0.f, sq = 0.f;
#pragma unroll
    for (int ii = 0; ii < 20; ++ii) {
      int c = lane + ii * 32;
      float v = obuf[j * 640 + c] + X[(size_t)n * 640 + c];
      vbuf[ii] = v;
      sum += v;
      sq += v * v;
    }
    for (int off = 16; off >= 1; off >>= 1) {
      sum += __shfl_xor(sum, off, 32);
      sq += __shfl_xor(sq, off, 32);
    }
    float mu = sum * (1.f / 640.f);
    float var = sq * (1.f / 640.f) - mu * mu;
    float rstd = rsqrtf(var + 1e-5f);
#pragma unroll
    for (int ii = 0; ii < 20; ++ii) {
      int c = lane + ii * 32;
      out[(size_t)n * 640 + c] = (vbuf[ii] - mu) * rstd * gamma[c] + beta[c];
    }
  }
}

// ---------------- launcher ----------------
extern "C" void kernel_launch(void* const* d_in, const int* in_sizes, int n_in,
                              void* d_out, int out_size, void* d_ws, size_t ws_size,
                              hipStream_t stream) {
  const float* x = (const float*)d_in[0];
  const float* Wq = (const float*)d_in[1];
  const float* Wk = (const float*)d_in[2];
  const float* Wv = (const float*)d_in[3];
  const float* gamma = (const float*)d_in[4];
  const float* beta = (const float*)d_in[5];
  float* out = (float*)d_out;

  const size_t NELEM = (size_t)25600 * 640;
  unsigned short* Qb = (unsigned short*)d_ws;
  unsigned short* Kb = Qb + NELEM;
  unsigned short* Vb = Kb + NELEM;

  qkv_gemm_kernel<<<dim3(200, 10, 3), 256, 0, stream>>>(x, Wq, Wk, Wv, Qb, Kb, Vb);

  size_t smem_bytes = (size_t)(400 * LDQ + 400 * LDQ + 64 * LDV + 400 * LDP) * 2 +
                      800 * 4;  // = 254,848 B (within 320 KB/WGP)
  attn_ln_kernel<<<dim3(640), 256, smem_bytes, stream>>>(Qb, Kb, Vb, x, gamma, beta,
                                                         out);
}
